// NDRM1_3272765080317
// MI455X (gfx1250) — matrix operand order repaired and checked by reading, not compiled, not run on hardware
//
#include <hip/hip_runtime.h>

// ---------------- model dimensions ----------------
#define B_   16
#define ND_  2
#define Q_   16
#define D_   512
#define H_   256
#define NH_  8
#define HD_  32
#define L_   2
#define K_   128
#define DP_  255                 // pooled length (512-4)/2+1
#define ROWS_D (B_*ND_*D_)       // 16384 doc-token rows
#define ROWS_Q (B_*Q_)           // 256 query rows

typedef unsigned short u16;
typedef __attribute__((ext_vector_type(16))) __bf16 v16bf;
typedef __attribute__((ext_vector_type(8)))  float  v8f;

union FragB16 { v16bf v; uint4 q[2]; u16 h[16]; };
union FragAcc { v8f v; float f[8]; };

__device__ __forceinline__ u16 f2bf(float f) {
  unsigned int u = __float_as_uint(f);
  unsigned int r = u + 0x7FFFu + ((u >> 16) & 1u);   // RNE
  return (u16)(r >> 16);
}

__device__ __forceinline__ v8f wmma_bf16(v16bf a, v16bf b, v8f c) {
  return __builtin_amdgcn_wmma_f32_16x16x32_bf16(false, a, false, b, (short)0, c,
                                                 false, false);
}

__device__ __forceinline__ float wave_sum(float v) {
#pragma unroll
  for (int m = 16; m >= 1; m >>= 1) v += __shfl_xor(v, m, 32);
  return v;
}
__device__ __forceinline__ float wave_max(float v) {
#pragma unroll
  for (int m = 16; m >= 1; m >>= 1) v = fmaxf(v, __shfl_xor(v, m, 32));
  return v;
}

// CDNA5 async global->LDS copy (16B per lane), tracked by ASYNCcnt.
// ldsOff: byte offset in workgroup LDS (low 32 bits of generic shared pointer).
// gOff:   byte offset added to the 64-bit SGPR base (GVS mode).
__device__ __forceinline__ void async_copy_b128(unsigned int ldsOff,
                                                unsigned int gOff,
                                                const void* base) {
  asm volatile("global_load_async_to_lds_b128 %0, %1, %2"
               :: "v"(ldsOff), "v"(gOff), "s"(base) : "memory");
}
__device__ __forceinline__ void wait_async0() {
  asm volatile("s_wait_asynccnt 0" ::: "memory");
}

// ---------------- small utility kernels ----------------
__global__ __launch_bounds__(256) void f32_to_bf16_kernel(const float* __restrict__ s,
                                                          u16* __restrict__ d, int n) {
  int i = blockIdx.x * 256 + threadIdx.x;
  if (i < n) d[i] = f2bf(s[i]);
}

// rbf_fc1_w (128x256 row-major, KxN) -> transposed bf16 (col n major, stride 128)
__global__ __launch_bounds__(256) void w1_transpose_kernel(const float* __restrict__ src,
                                                           u16* __restrict__ dst) {
  int i = blockIdx.x * 256 + threadIdx.x;    // 0..32767
  int k = i >> 8, n = i & 255;
  dst[n * 128 + k] = f2bf(src[i]);
}

__global__ __launch_bounds__(256) void embed_q_kernel(const int* __restrict__ tok,
                                                      const float* __restrict__ emb,
                                                      u16* __restrict__ qe) {
  int row = blockIdx.x, c = threadIdx.x;
  int t = tok[row];
  qe[(size_t)row * H_ + c] = f2bf(emb[(size_t)t * H_ + c]);
}

__global__ __launch_bounds__(256) void embed_d_kernel(const int* __restrict__ tok,
                                                      const float* __restrict__ emb,
                                                      float* __restrict__ de,
                                                      float* __restrict__ x,
                                                      u16* __restrict__ xb) {
  int row = blockIdx.x, c = threadIdx.x;
  int t = tok[row];
  float e = emb[(size_t)t * H_ + c];
  int s = row & (D_ - 1);
  // sinusoidal PE: arg = pos * exp(-(c&~1)*ln(10000)/H)
  float arg = (float)s * __expf((float)(c & ~1) * (-9.210340371976184f / (float)H_));
  float pe = (c & 1) ? __cosf(arg) : __sinf(arg);
  size_t o = (size_t)row * H_ + c;
  de[o] = e;
  float xv = e + pe;
  x[o] = xv;
  xb[o] = f2bf(xv);
}

// ---------------- generic WMMA GEMM: C = A(MxK) @ B(KxN) + bias ----------------
// block tile 128x128, 8 waves, each wave 64x32 (4x2 fragments), K step 32.
// A tile is staged with CDNA5 async global->LDS b128 copies.
template <bool OUTF, bool OUTB, bool RELU>
__global__ __launch_bounds__(256) void gemm_wmma_kernel(
    const u16* __restrict__ A, const u16* __restrict__ Bm,
    const float* __restrict__ bias, float* __restrict__ outF,
    u16* __restrict__ outB, int M, int N, int Kd) {
  __shared__ __align__(16) u16 sA[128 * 32];   // [row][k]
  __shared__ __align__(16) u16 sBt[128 * 32];  // [col][k]
  const int tid = threadIdx.x;
  const int lane = tid & 31, wave = tid >> 5;
  const int half = lane >> 4, idx = lane & 15;
  const int wm = wave >> 2, wn = wave & 3;
  const int bm = blockIdx.y * 128, bn = blockIdx.x * 128;

  FragAcc acc[4][2];
#pragma unroll
  for (int mi = 0; mi < 4; ++mi)
#pragma unroll
    for (int ni = 0; ni < 2; ++ni)
#pragma unroll
      for (int i = 0; i < 8; ++i) acc[mi][ni].f[i] = 0.f;

  for (int k0 = 0; k0 < Kd; k0 += 32) {
#pragma unroll
    for (int v = 0; v < 2; ++v) {   // A tile: 512 x 16B, async direct to LDS
      int vi = tid * 2 + v;
      int row = vi >> 2, colv = (vi & 3) << 3;
      unsigned int ldsOff = (unsigned int)(uintptr_t)(sA + row * 32 + colv);
      unsigned int gOff = (unsigned int)(((size_t)(bm + row) * Kd + k0 + colv) * 2);
      async_copy_b128(ldsOff, gOff, A);
    }
#pragma unroll
    for (int v = 0; v < 2; ++v) {   // B tile, transpose through VGPRs
      int vi = tid * 2 + v;
      int kr = vi >> 4, c0 = (vi & 15) << 3;
      uint4 dv = *(const uint4*)(Bm + (size_t)(k0 + kr) * N + bn + c0);
      const u16* ds = (const u16*)&dv;
#pragma unroll
      for (int j = 0; j < 8; ++j) sBt[(c0 + j) * 32 + kr] = ds[j];
    }
    wait_async0();
    __syncthreads();
    FragB16 af[4], bw[2];
#pragma unroll
    for (int mi = 0; mi < 4; ++mi) {
      int r = wm * 64 + mi * 16 + idx;
      af[mi].q[0] = *(const uint4*)(sA + r * 32 + 16 * half);
      af[mi].q[1] = *(const uint4*)(sA + r * 32 + 16 * half + 8);
    }
#pragma unroll
    for (int ni = 0; ni < 2; ++ni) {
      int c = wn * 32 + ni * 16 + idx;
      bw[ni].q[0] = *(const uint4*)(sBt + c * 32 + 16 * half);
      bw[ni].q[1] = *(const uint4*)(sBt + c * 32 + 16 * half + 8);
    }
#pragma unroll
    for (int mi = 0; mi < 4; ++mi)
#pragma unroll
      for (int ni = 0; ni < 2; ++ni)
        acc[mi][ni].v = wmma_bf16(af[mi].v, bw[ni].v, acc[mi][ni].v);
    __syncthreads();
  }
#pragma unroll
  for (int mi = 0; mi < 4; ++mi)
#pragma unroll
    for (int ni = 0; ni < 2; ++ni) {
      int n = bn + wn * 32 + ni * 16 + idx;
      float bv = bias[n];
#pragma unroll
      for (int i = 0; i < 8; ++i) {
        int m = bm + wm * 64 + mi * 16 + i + 8 * half;
        float v = acc[mi][ni].f[i] + bv;
        if (RELU) v = fmaxf(v, 0.f);
        size_t o = (size_t)m * N + n;
        if (OUTF) outF[o] = v;
        if (OUTB) outB[o] = f2bf(v);
      }
    }
}

// ---------------- fused attention per (seq bn, head h, 16-query block) ----------------
// whole-head K and V staged LDS-resident via async b128 copies (320KB WGP LDS).
__global__ __launch_bounds__(256) void attn_kernel(const u16* __restrict__ qkv,
                                                   const float* __restrict__ mask_d,
                                                   u16* __restrict__ o) {
  extern __shared__ __align__(16) unsigned char smem[];
  u16* sQ = (u16*)smem;                 // 16x32
  u16* sK = sQ + 16 * 32;               // 512x32
  u16* sV = sK + 512 * 32;              // 512x32
  float* sS = (float*)(sV + 512 * 32);  // 16x512
  u16* sP = (u16*)(sS + 16 * 512);      // 16x512

  const int tid = threadIdx.x;
  const int lane = tid & 31, wave = tid >> 5;
  const int half = lane >> 4, idx = lane & 15;
  const int qb = blockIdx.x, h = blockIdx.y, bn = blockIdx.z;
  const size_t rowBase = (size_t)bn * D_;

  for (int i = tid; i < 2048; i += 256) {   // whole K and V for head, async
    int r = i >> 2, cv = (i & 3) << 3;
    unsigned int gK = (unsigned int)((((rowBase + r) * 768) + 256 + h * 32 + cv) * 2);
    unsigned int gV = (unsigned int)((((rowBase + r) * 768) + 512 + h * 32 + cv) * 2);
    async_copy_b128((unsigned int)(uintptr_t)(sK + r * 32 + cv), gK, qkv);
    async_copy_b128((unsigned int)(uintptr_t)(sV + r * 32 + cv), gV, qkv);
  }
  for (int i = tid; i < 64; i += 256) {     // Q block (tiny, regular path)
    int r = i >> 2, cv = (i & 3) << 3;
    *(uint4*)(sQ + r * 32 + cv) =
        *(const uint4*)(qkv + (rowBase + qb * 16 + r) * 768 + h * 32 + cv);
  }
  wait_async0();
  __syncthreads();

  FragB16 aQ;
  aQ.q[0] = *(const uint4*)(sQ + idx * 32 + 16 * half);
  aQ.q[1] = *(const uint4*)(sQ + idx * 32 + 16 * half + 8);
  const float scale = 0.1767766952966369f;  // 1/sqrt(32)
  for (int nt = wave; nt < 32; nt += 8) {   // 32 key tiles of 16
    FragB16 bK;
    int key = nt * 16 + idx;
    bK.q[0] = *(const uint4*)(sK + key * 32 + 16 * half);
    bK.q[1] = *(const uint4*)(sK + key * 32 + 16 * half + 8);
    FragAcc c;
#pragma unroll
    for (int i = 0; i < 8; ++i) c.f[i] = 0.f;
    c.v = wmma_bf16(aQ.v, bK.v, c.v);
    float mk = mask_d[rowBase + key];
#pragma unroll
    for (int i = 0; i < 8; ++i) {
      int m = i + 8 * half;
      sS[m * 512 + key] = (mk > 0.f) ? c.f[i] * scale : -1e9f;
    }
  }
  __syncthreads();

  for (int r = wave * 2; r < wave * 2 + 2; ++r) {       // softmax, 2 rows/wave
    float mx = -1e30f;
    for (int cc = lane; cc < 512; cc += 32) mx = fmaxf(mx, sS[r * 512 + cc]);
    mx = wave_max(mx);
    float sum = 0.f;
    for (int cc = lane; cc < 512; cc += 32) sum += __expf(sS[r * 512 + cc] - mx);
    sum = wave_sum(sum);
    float inv = 1.f / sum;
    for (int cc = lane; cc < 512; cc += 32)
      sP[r * 512 + cc] = f2bf(__expf(sS[r * 512 + cc] - mx) * inv);
  }
  __syncthreads();

  if (wave < 2) {                                       // O = P @ V, 2 hd tiles
    int nt = wave;
    FragAcc c;
#pragma unroll
    for (int i = 0; i < 8; ++i) c.f[i] = 0.f;
    for (int kc = 0; kc < 16; ++kc) {
      FragB16 aP, bV;
      aP.q[0] = *(const uint4*)(sP + idx * 512 + kc * 32 + 16 * half);
      aP.q[1] = *(const uint4*)(sP + idx * 512 + kc * 32 + 16 * half + 8);
#pragma unroll
      for (int e = 0; e < 16; ++e) {
        int k = kc * 32 + e + 16 * half;
        bV.h[e] = sV[k * 32 + nt * 16 + idx];
      }
      c.v = wmma_bf16(aP.v, bV.v, c.v);
    }
#pragma unroll
    for (int i = 0; i < 8; ++i) {
      int m = i + 8 * half;
      o[(rowBase + qb * 16 + m) * 256 + h * 32 + nt * 16 + idx] = f2bf(c.f[i]);
    }
  }
}

// ---------------- residual + LayerNorm (wave per row) ----------------
__global__ __launch_bounds__(256) void resln_kernel(const float* __restrict__ tmp,
                                                    float* __restrict__ x,
                                                    u16* __restrict__ xb,
                                                    const float* __restrict__ g,
                                                    const float* __restrict__ b) {
  int wid = blockIdx.x * 8 + (threadIdx.x >> 5);
  int lane = threadIdx.x & 31;
  const float* xr = x + (size_t)wid * H_;
  const float* tr = tmp + (size_t)wid * H_;
  float v[8];
  float s = 0.f;
#pragma unroll
  for (int i = 0; i < 8; ++i) {
    v[i] = xr[lane + i * 32] + tr[lane + i * 32];
    s += v[i];
  }
  s = wave_sum(s);
  float mean = s * (1.f / H_);
  float var = 0.f;
#pragma unroll
  for (int i = 0; i < 8; ++i) { float d = v[i] - mean; var += d * d; }
  var = wave_sum(var) * (1.f / H_);
  float inv = rsqrtf(var + 1e-5f);
#pragma unroll
  for (int i = 0; i < 8; ++i) {
    int c = lane + i * 32;
    float ov = (v[i] - mean) * inv * g[c] + b[c];
    x[(size_t)wid * H_ + c] = ov;
    xb[(size_t)wid * H_ + c] = f2bf(ov);
  }
}

// ---------------- d_c = w0*d_e + w1*d_ctx + b; row norms ----------------
__global__ __launch_bounds__(256) void dctx_kernel(const float* __restrict__ de,
                                                   const float* __restrict__ x,
                                                   const float* __restrict__ wc,
                                                   const float* __restrict__ bc,
                                                   float* __restrict__ dcO,
                                                   float* __restrict__ dn) {
  int wid = blockIdx.x * 8 + (threadIdx.x >> 5);
  int lane = threadIdx.x & 31;
  float w0 = wc[0], w1v = wc[1], bv = bc[0];
  float s = 0.f;
#pragma unroll
  for (int i = 0; i < 8; ++i) {
    size_t o = (size_t)wid * H_ + lane + i * 32;
    float v = w0 * de[o] + w1v * x[o] + bv;
    dcO[o] = v;
    s += v * v;
  }
  s = wave_sum(s);
  if (lane == 0) dn[wid] = fmaxf(sqrtf(s), 1e-8f);
}

__global__ __launch_bounds__(256) void qnorm_kernel(const float* __restrict__ qt,
                                                    float* __restrict__ qn) {
  int wid = blockIdx.x * 8 + (threadIdx.x >> 5);
  int lane = threadIdx.x & 31;
  float s = 0.f;
#pragma unroll
  for (int i = 0; i < 8; ++i) {
    float v = qt[(size_t)wid * H_ + lane + i * 32];
    s += v * v;
  }
  s = wave_sum(s);
  if (lane == 0) qn[wid] = fmaxf(sqrtf(s), 1e-8f);
}

// ---------------- cosine matrix (B,ND,Q,D); wave per (bn,d) ----------------
__global__ __launch_bounds__(256) void cos_kernel(const float* __restrict__ qt,
                                                  const float* __restrict__ qn,
                                                  const float* __restrict__ dc,
                                                  const float* __restrict__ dn,
                                                  float* __restrict__ cosM) {
  int wid = blockIdx.x * 8 + (threadIdx.x >> 5);  // 0..16383 = bn*512+d
  int lane = threadIdx.x & 31;
  int bn = wid >> 9, dpos = wid & 511;
  int b = bn >> 1;
  float dv[8];
#pragma unroll
  for (int i = 0; i < 8; ++i) dv[i] = dc[(size_t)wid * H_ + lane + i * 32];
  float dnv = dn[wid];
  for (int qq = 0; qq < Q_; ++qq) {
    const float* qr = qt + (size_t)(b * Q_ + qq) * H_;
    float s = 0.f;
#pragma unroll
    for (int i = 0; i < 8; ++i) s += dv[i] * qr[lane + i * 32];
    s = wave_sum(s);
    if (lane == 0)
      cosM[(size_t)(bn * Q_ + qq) * D_ + dpos] = s / (qn[b * Q_ + qq] * dnv);
  }
}

// ---------------- fused RBF pooling + FC1(relu) + FC2 + max ----------------
__global__ __launch_bounds__(256) void rbf_kernel(
    const float* __restrict__ cosM, const float* __restrict__ mask_d,
    const float* __restrict__ mask_q, const u16* __restrict__ w1t,  // [n*128+k]
    const float* __restrict__ b1, const float* __restrict__ w2,
    const float* __restrict__ b2, float* __restrict__ sQbuf) {
  __shared__ __align__(16) u16 sY[16 * 128];
  __shared__ float sRow[16];
  __shared__ float sBest[16];
  const int tid = threadIdx.x;
  const int lane = tid & 31, wave = tid >> 5;
  const int half = lane >> 4, idx = lane & 15;
  const int bnq = blockIdx.x;
  const int qi = bnq & 15, bn = bnq >> 4, b = bn >> 1;
  const float* cosRow = cosM + (size_t)bnq * D_;
  const float* mrow = mask_d + (size_t)bn * D_;
  float best = -1e30f;

  for (int dpb = 0; dpb < 16; ++dpb) {
    for (int i = tid; i < 2048; i += 256) {  // build y tile (16 dp x 128 k)
      int r = i >> 7, k = i & 127;
      int dp = dpb * 16 + r;
      float y = 0.f;
      if (dp < DP_) {
        float mu = 1.f - (float)k * (1.f / 64.f);
        float pooled = 0.f;
#pragma unroll
        for (int j = 0; j < 4; ++j) {
          int dd = 2 * dp + j;
          float df = cosRow[dd] - mu;
          pooled += mrow[dd] * __expf(-df * df * 50.f);  // 1/(2*0.01)
        }
        y = __logf(pooled + 1e-6f);
      }
      sY[r * 128 + k] = f2bf(y);
    }
    if (tid < 16) sRow[tid] = b2[0];
    __syncthreads();

    float part[8];
#pragma unroll
    for (int i = 0; i < 8; ++i) part[i] = 0.f;
#pragma unroll
    for (int t2 = 0; t2 < 2; ++t2) {  // 2 n-tiles of 16 per wave -> 256 cols
      int nt = wave * 2 + t2;
      FragAcc c;
#pragma unroll
      for (int i = 0; i < 8; ++i) c.f[i] = 0.f;
#pragma unroll
      for (int kc = 0; kc < 4; ++kc) {
        FragB16 a, bw;
        a.q[0] = *(const uint4*)(sY + idx * 128 + kc * 32 + 16 * half);
        a.q[1] = *(const uint4*)(sY + idx * 128 + kc * 32 + 16 * half + 8);
        int col = nt * 16 + idx;
        bw.q[0] = *(const uint4*)(w1t + col * 128 + kc * 32 + 16 * half);
        bw.q[1] = *(const uint4*)(w1t + col * 128 + kc * 32 + 16 * half + 8);
        c.v = wmma_bf16(a.v, bw.v, c.v);
      }
      int ncol = nt * 16 + idx;
      float bb = b1[ncol], ww = w2[ncol];
#pragma unroll
      for (int i = 0; i < 8; ++i) {
        float hv = fmaxf(c.f[i] + bb, 0.f);
        part[i] += hv * ww;
      }
    }
#pragma unroll
    for (int i = 0; i < 8; ++i) {  // reduce over 16 cols per half-group
      float p = part[i];
#pragma unroll
      for (int m = 8; m >= 1; m >>= 1) p += __shfl_xor(p, m, 32);
      if (idx == 0) atomicAdd(&sRow[i + 8 * half], p);
    }
    __syncthreads();
    if (tid < 16) {
      int dp = dpb * 16 + tid;
      if (dp < DP_) best = fmaxf(best, sRow[tid]);
    }
    __syncthreads();
  }
  if (tid < 16) sBest[tid] = best;
  __syncthreads();
  if (tid == 0) {
    float m = -1e30f;
#pragma unroll
    for (int i = 0; i < 16; ++i) m = fmaxf(m, sBest[i]);
    sQbuf[bnq] = m * mask_q[b * Q_ + qi];
  }
}

__global__ void final_sum_kernel(const float* __restrict__ sQbuf,
                                 float* __restrict__ out) {
  int t = threadIdx.x;  // 0..31 == b*ND+n
  if (t < B_ * ND_) {
    float s = 0.f;
    for (int qq = 0; qq < Q_; ++qq) s += sQbuf[t * Q_ + qq];
    out[t] = s;
  }
}

// ---------------- host launcher ----------------
extern "C" void kernel_launch(void* const* d_in, const int* in_sizes, int n_in,
                              void* d_out, int out_size, void* d_ws, size_t ws_size,
                              hipStream_t stream) {
  (void)in_sizes; (void)n_in; (void)out_size; (void)ws_size;
  const int*   qTok   = (const int*)d_in[0];
  const int*   dTok   = (const int*)d_in[1];
  const float* mask_q = (const float*)d_in[2];
  const float* mask_d = (const float*)d_in[3];
  const float* embed  = (const float*)d_in[4];
  const float* fcqt_w = (const float*)d_in[5];
  const float* fcqt_b = (const float*)d_in[6];
  const float* qkv_w  = (const float*)d_in[7];
  const float* qkv_b  = (const float*)d_in[8];
  const float* out_w  = (const float*)d_in[9];
  const float* out_b  = (const float*)d_in[10];
  const float* ln1_g  = (const float*)d_in[11];
  const float* ln1_b  = (const float*)d_in[12];
  const float* f1_w   = (const float*)d_in[13];
  const float* f1_b   = (const float*)d_in[14];
  const float* f2_w   = (const float*)d_in[15];
  const float* f2_b   = (const float*)d_in[16];
  const float* ln2_g  = (const float*)d_in[17];
  const float* ln2_b  = (const float*)d_in[18];
  const float* ctx_w  = (const float*)d_in[19];
  const float* ctx_b  = (const float*)d_in[20];
  const float* r1_w   = (const float*)d_in[21];
  const float* r1_b   = (const float*)d_in[22];
  const float* r2_w   = (const float*)d_in[23];
  const float* r2_b   = (const float*)d_in[24];
  float* out = (float*)d_out;

  char* p = (char*)d_ws;
  auto alloc = [&](size_t bytes) {
    char* r = p;
    p += (bytes + 255) & ~(size_t)255;
    return r;
  };
  u16* w_fcqt = (u16*)alloc((size_t)H_ * H_ * 2);
  u16* w_qkv  = (u16*)alloc((size_t)L_ * H_ * 3 * H_ * 2);
  u16* w_out  = (u16*)alloc((size_t)L_ * H_ * H_ * 2);
  u16* w_f1   = (u16*)alloc((size_t)L_ * H_ * H_ * 2);
  u16* w_f2   = (u16*)alloc((size_t)L_ * H_ * H_ * 2);
  u16* w_r1t  = (u16*)alloc((size_t)K_ * H_ * 2);
  u16* qe_bf  = (u16*)alloc((size_t)ROWS_Q * H_ * 2);
  float* qt   = (float*)alloc((size_t)ROWS_Q * H_ * 4);
  float* de   = (float*)alloc((size_t)ROWS_D * H_ * 4);
  float* xf   = (float*)alloc((size_t)ROWS_D * H_ * 4);
  u16* xbf    = (u16*)alloc((size_t)ROWS_D * H_ * 2);
  u16* qkvbf  = (u16*)alloc((size_t)ROWS_D * 3 * H_ * 2);
  u16* obf    = (u16*)alloc((size_t)ROWS_D * H_ * 2);
  float* tmpf = (float*)alloc((size_t)ROWS_D * H_ * 4);
  u16* ffh    = (u16*)alloc((size_t)ROWS_D * H_ * 2);
  float* dc   = (float*)alloc((size_t)ROWS_D * H_ * 4);
  float* dnrm = (float*)alloc((size_t)ROWS_D * 4);
  float* qnrm = (float*)alloc((size_t)ROWS_Q * 4);
  float* cosM = (float*)alloc((size_t)B_ * ND_ * Q_ * D_ * 4);
  float* sQb  = (float*)alloc((size_t)B_ * ND_ * Q_ * 4);

  auto cvt = [&](const float* s, u16* dst, int n) {
    f32_to_bf16_kernel<<<dim3((n + 255) / 256), dim3(256), 0, stream>>>(s, dst, n);
  };
  cvt(fcqt_w, w_fcqt, H_ * H_);
  cvt(qkv_w, w_qkv, L_ * H_ * 3 * H_);
  cvt(out_w, w_out, L_ * H_ * H_);
  cvt(f1_w, w_f1, L_ * H_ * H_);
  cvt(f2_w, w_f2, L_ * H_ * H_);
  w1_transpose_kernel<<<dim3(128), dim3(256), 0, stream>>>(r1_w, w_r1t);

  embed_q_kernel<<<dim3(ROWS_Q), dim3(256), 0, stream>>>(qTok, embed, qe_bf);
  gemm_wmma_kernel<true, false, false>
      <<<dim3(H_ / 128, ROWS_Q / 128), dim3(256), 0, stream>>>(
          qe_bf, w_fcqt, fcqt_b, qt, (u16*)nullptr, ROWS_Q, H_, H_);

  embed_d_kernel<<<dim3(ROWS_D), dim3(256), 0, stream>>>(dTok, embed, de, xf, xbf);

  size_t attnLds = (size_t)(16 * 32 + 512 * 32 + 512 * 32) * 2 +
                   (size_t)16 * 512 * 4 + (size_t)16 * 512 * 2;  // 115712 B
  for (int l = 0; l < L_; ++l) {
    gemm_wmma_kernel<false, true, false>
        <<<dim3((3 * H_) / 128, ROWS_D / 128), dim3(256), 0, stream>>>(
            xbf, w_qkv + (size_t)l * H_ * 3 * H_, qkv_b + (size_t)l * 3 * H_,
            (float*)nullptr, qkvbf, ROWS_D, 3 * H_, H_);
    attn_kernel<<<dim3(32, 8, 32), dim3(256), attnLds, stream>>>(qkvbf, mask_d, obf);
    gemm_wmma_kernel<true, false, false>
        <<<dim3(H_ / 128, ROWS_D / 128), dim3(256), 0, stream>>>(
            obf, w_out + (size_t)l * H_ * H_, out_b + (size_t)l * H_, tmpf,
            (u16*)nullptr, ROWS_D, H_, H_);
    resln_kernel<<<dim3(ROWS_D / 8), dim3(256), 0, stream>>>(
        tmpf, xf, xbf, ln1_g + l * H_, ln1_b + l * H_);
    gemm_wmma_kernel<false, true, true>
        <<<dim3(H_ / 128, ROWS_D / 128), dim3(256), 0, stream>>>(
            xbf, w_f1 + (size_t)l * H_ * H_, f1_b + (size_t)l * H_,
            (float*)nullptr, ffh, ROWS_D, H_, H_);
    gemm_wmma_kernel<true, false, false>
        <<<dim3(H_ / 128, ROWS_D / 128), dim3(256), 0, stream>>>(
            ffh, w_f2 + (size_t)l * H_ * H_, f2_b + (size_t)l * H_, tmpf,
            (u16*)nullptr, ROWS_D, H_, H_);
    resln_kernel<<<dim3(ROWS_D / 8), dim3(256), 0, stream>>>(
        tmpf, xf, xbf, ln2_g + l * H_, ln2_b + l * H_);
  }

  dctx_kernel<<<dim3(ROWS_D / 8), dim3(256), 0, stream>>>(de, xf, ctx_w, ctx_b, dc, dnrm);
  qnorm_kernel<<<dim3(ROWS_Q / 8), dim3(256), 0, stream>>>(qt, qnrm);
  cos_kernel<<<dim3(ROWS_D / 8), dim3(256), 0, stream>>>(qt, qnrm, dc, dnrm, cosM);
  rbf_kernel<<<dim3(B_ * ND_ * Q_), dim3(256), 0, stream>>>(cosM, mask_d, mask_q,
                                                            w_r1t, r1_b, r2_w, r2_b, sQb);
  final_sum_kernel<<<dim3(1), dim3(32), 0, stream>>>(sQb, out);
}
